// NodeProcessor_31825707663673
// MI455X (gfx1250) — matrix operand order repaired and checked by reading, not compiled, not run on hardware
//
#include <hip/hip_runtime.h>
#include <hip/hip_fp16.h>

#define N_NODES 50000
#define N_EDGES 800000
#define D       128
#define TWO_D   256

typedef __attribute__((ext_vector_type(16))) _Float16 v16h;
typedef __attribute__((ext_vector_type(8)))  _Float16 v8h;
typedef __attribute__((ext_vector_type(4)))  _Float16 v4h;
typedef __attribute__((ext_vector_type(8)))  float    v8f;
typedef __attribute__((ext_vector_type(4)))  float    v4f;

// ---------------------------------------------------------------- zero agg
__global__ void k_zero(float4* __restrict__ agg4, int n4) {
    int i = blockIdx.x * blockDim.x + threadIdx.x;
    if (i < n4) agg4[i] = make_float4(0.f, 0.f, 0.f, 0.f);
}

// ---------------------- weights: f32 row-major -> f16 transposed (Wt[n][k])
__global__ void k_prep(const float* __restrict__ W1, const float* __restrict__ W2,
                       _Float16* __restrict__ W1t, _Float16* __restrict__ W2t) {
    int i = blockIdx.x * blockDim.x + threadIdx.x;
    if (i < TWO_D * TWO_D) {                      // W1: [256 k][256 n] -> W1t[n][k]
        int n = i / TWO_D, k = i % TWO_D;
        W1t[n * TWO_D + k] = (_Float16)W1[k * TWO_D + n];
    } else if (i < TWO_D * TWO_D + D * TWO_D) {   // W2: [256 k][128 n] -> W2t[n][k]
        int j = i - TWO_D * TWO_D;
        int n = j / TWO_D, k = j % TWO_D;
        W2t[n * TWO_D + k] = (_Float16)W2[k * D + n];
    }
}

// ------------- edge scatter: NT streaming loads + native f32 atomics into L2-resident agg
__global__ void k_scatter(const float* __restrict__ edge_attr,
                          const int*   __restrict__ edge_dst,
                          float*       __restrict__ agg) {
    long long gid = (long long)blockIdx.x * blockDim.x + threadIdx.x;
    int e = (int)(gid >> 5);
    if (e >= N_EDGES) return;
    int c = ((int)gid & 31) * 4;
    int dst = edge_dst[e];
    // edge_attr is streamed exactly once (410 MB): non-temporal so it doesn't
    // evict the 25.6 MB agg accumulator from the 192 MB L2.
    v4f v = __builtin_nontemporal_load((const v4f*)(edge_attr + (size_t)e * D + c));
    float* a = agg + (size_t)dst * D + c;
    unsafeAtomicAdd(a + 0, v[0]);   // global_atomic_add_f32
    unsafeAtomicAdd(a + 1, v[1]);
    unsafeAtomicAdd(a + 2, v[2]);
    unsafeAtomicAdd(a + 3, v[3]);
}

// A-fragment (contiguous k order, same as B): lanes 0-15 take K[k0..k0+16),
// lanes 16-31 take K[k0+16..k0+32)  -> two 16B LDS loads.
__device__ __forceinline__ v16h load_afrag(const _Float16* buf, int m, int sel, int k0) {
    const _Float16* p = buf + m * TWO_D + k0 + sel * 16;
    v8h lo = *(const v8h*)(p);
    v8h hi = *(const v8h*)(p + 8);
    return __builtin_shufflevector(lo, hi, 0,1,2,3,4,5,6,7,8,9,10,11,12,13,14,15);
}

// ----------------------------- fused concat -> GEMM1+SiLU -> GEMM2 -> LN -> residual
__global__ void __launch_bounds__(128)
k_mlp(const float* __restrict__ x,    const float* __restrict__ agg,
      const _Float16* __restrict__ W1t, const float* __restrict__ b1,
      const _Float16* __restrict__ W2t, const float* __restrict__ b2,
      const float* __restrict__ gamma, const float* __restrict__ beta,
      float* __restrict__ out) {
    __shared__ __align__(16) _Float16 sX[4][16 * TWO_D];   // [x|agg] tile (reused as f32 OUT)
    __shared__ __align__(16) _Float16 sH[4][16 * TWO_D];   // hidden tile

    const int wave = threadIdx.x >> 5;
    const int lane = threadIdx.x & 31;
    const int tile = blockIdx.x * 4 + wave;
    if (tile * 16 >= N_NODES) return;                      // no block barriers -> safe
    const int r0 = tile * 16;

    _Float16* bufX = sX[wave];
    _Float16* bufH = sH[wave];

    // ---- stage 0: stage [x | agg] as fp16, 16 rows x 256 cols
    for (int idx = lane; idx < 16 * 64; idx += 32) {       // float4 granules
        int row = idx >> 6;
        int col = (idx & 63) * 4;
        const float* src = (col < D) ? (x   + (size_t)(r0 + row) * D + col)
                                     : (agg + (size_t)(r0 + row) * D + (col - D));
        float4 v = *(const float4*)src;
        v4h h; h[0] = (_Float16)v.x; h[1] = (_Float16)v.y;
               h[2] = (_Float16)v.z; h[3] = (_Float16)v.w;
        *(v4h*)(bufX + row * TWO_D + col) = h;
    }

    const int m   = lane & 15;        // A row / B column within 16x16 tile
    const int sel = lane >> 4;        // K half selector

    // ---- stage 1: H = silu([x|agg] @ W1 + b1)   (16 N-tiles x 8 K-blocks of WMMA)
    for (int nt = 0; nt < 16; ++nt) {
        const int n0 = nt * 16;
        const float bv = b1[n0 + m];
        v8f acc;
        #pragma unroll
        for (int r = 0; r < 8; ++r) acc[r] = bv;
        const _Float16* bcol = W1t + (size_t)(n0 + m) * TWO_D + sel * 16;
        #pragma unroll
        for (int kb = 0; kb < 8; ++kb) {
            v16h a = load_afrag(bufX, m, sel, kb * 32);
            v16h b = *(const v16h*)(bcol + kb * 32);
            acc = __builtin_amdgcn_wmma_f32_16x16x32_f16(false, a, false, b,
                                                         (short)0, acc, false, false);
        }
        #pragma unroll
        for (int r = 0; r < 8; ++r) {
            float h = acc[r];
            // SiLU with v_exp_f32 + v_rcp_f32 (both TRANS: co-execute with WMMA pipe)
            float s = h * __builtin_amdgcn_rcpf(1.0f + __expf(-h));
            bufH[(r + 8 * sel) * TWO_D + n0 + m] = (_Float16)s;
        }
    }

    // ---- stage 2: OUT = H @ W2 + b2  (OUT f32 reuses bufX: 16x128)
    float* outT = (float*)bufX;
    for (int nt = 0; nt < 8; ++nt) {
        const int n0 = nt * 16;
        const float bv = b2[n0 + m];
        v8f acc;
        #pragma unroll
        for (int r = 0; r < 8; ++r) acc[r] = bv;
        const _Float16* bcol = W2t + (size_t)(n0 + m) * TWO_D + sel * 16;
        #pragma unroll
        for (int kb = 0; kb < 8; ++kb) {
            v16h a = load_afrag(bufH, m, sel, kb * 32);
            v16h b = *(const v16h*)(bcol + kb * 32);
            acc = __builtin_amdgcn_wmma_f32_16x16x32_f16(false, a, false, b,
                                                         (short)0, acc, false, false);
        }
        #pragma unroll
        for (int r = 0; r < 8; ++r)
            outT[(r + 8 * sel) * D + n0 + m] = acc[r];
    }

    // ---- stage 3: LayerNorm + residual. 2 lanes per row, 64 cols each.
    {
        const int row  = lane & 15;
        const int half = lane >> 4;
        const float* rp = outT + row * D + half * 64;
        float s = 0.f, ss = 0.f;
        #pragma unroll
        for (int c = 0; c < 64; c += 4) {
            float4 v = *(const float4*)(rp + c);
            s  += v.x + v.y + v.z + v.w;
            ss += v.x * v.x + v.y * v.y + v.z * v.z + v.w * v.w;
        }
        s  += __shfl_xor(s, 16, 32);   // combine the two half-row lanes (wave32)
        ss += __shfl_xor(ss, 16, 32);
        const float mean = s * (1.f / 128.f);
        const float var  = ss * (1.f / 128.f) - mean * mean;
        const float rstd = rsqrtf(var + 1e-5f);

        const float* xr   = x   + (size_t)(r0 + row) * D + half * 64;
        float*       orow = out + (size_t)(r0 + row) * D + half * 64;
        const float* g  = gamma + half * 64;
        const float* bb = beta  + half * 64;
        #pragma unroll
        for (int c = 0; c < 64; c += 4) {
            float4 h4 = *(const float4*)(rp + c);
            float4 x4 = *(const float4*)(xr + c);
            float4 g4 = *(const float4*)(g  + c);
            float4 b4 = *(const float4*)(bb + c);
            v4f o;
            o[0] = x4.x + (h4.x - mean) * rstd * g4.x + b4.x;
            o[1] = x4.y + (h4.y - mean) * rstd * g4.y + b4.y;
            o[2] = x4.z + (h4.z - mean) * rstd * g4.z + b4.z;
            o[3] = x4.w + (h4.w - mean) * rstd * g4.w + b4.w;
            __builtin_nontemporal_store(o, (v4f*)(orow + c));  // write-once output
        }
    }
}

extern "C" void kernel_launch(void* const* d_in, const int* in_sizes, int n_in,
                              void* d_out, int out_size, void* d_ws, size_t ws_size,
                              hipStream_t stream) {
    const float* x          = (const float*)d_in[0];
    const int*   edge_index = (const int*)  d_in[1];   // (2, E): row 1 = dest
    const float* edge_attr  = (const float*)d_in[2];
    const float* W1         = (const float*)d_in[3];
    const float* b1         = (const float*)d_in[4];
    const float* W2         = (const float*)d_in[5];
    const float* b2         = (const float*)d_in[6];
    const float* ln_gamma   = (const float*)d_in[7];
    const float* ln_beta    = (const float*)d_in[8];
    float* out = (float*)d_out;

    // workspace: agg (25.6 MB f32) | W1t fp16 (128 KB) | W2t fp16 (64 KB)
    float*     agg = (float*)d_ws;
    size_t     aggBytes = (size_t)N_NODES * D * sizeof(float);
    _Float16*  W1t = (_Float16*)((char*)d_ws + aggBytes);
    _Float16*  W2t = W1t + TWO_D * TWO_D;

    int n4 = N_NODES * D / 4;
    k_zero<<<(n4 + 255) / 256, 256, 0, stream>>>((float4*)agg, n4);

    int prepN = TWO_D * TWO_D + D * TWO_D;
    k_prep<<<(prepN + 255) / 256, 256, 0, stream>>>(W1, W2, W1t, W2t);

    long long nthr = (long long)N_EDGES * 32;
    k_scatter<<<(unsigned)((nthr + 255) / 256), 256, 0, stream>>>(
        edge_attr, edge_index + N_EDGES, agg);

    int ntiles  = N_NODES / 16;          // 3125
    int nblocks = (ntiles + 3) / 4;      // 4 waves / block, 1 tile / wave
    k_mlp<<<nblocks, 128, 0, stream>>>(x, agg, W1t, b1, W2t, b2,
                                       ln_gamma, ln_beta, out);
}
